// PLinear_Complex_70987219468443
// MI455X (gfx1250) — matrix operand order repaired and checked by reading, not compiled, not run on hardware
//
#include <hip/hip_runtime.h>

// Problem dims (from reference): x: [8192, 4096], w: [4096, 4096], y: [8192, 4096] x2
#define NROWS 8192
#define KDIM  4096
#define ODIM  4096

// Tiling
#define BM 128          // output rows per block
#define BN 128          // output cols per block
#define BK 32           // K per step (one wmma_16x16x32 depth)
#define KSTEPS (KDIM / BK)
#define LDS_STRIDE 40   // bf16 elems per LDS row: 32 data + 8 pad (80B stride -> conflict-free)

typedef __bf16 bf16;
typedef __attribute__((ext_vector_type(4)))  __bf16 v4bf;
typedef __attribute__((ext_vector_type(8)))  __bf16 v8bf;
typedef __attribute__((ext_vector_type(16))) __bf16 v16bf;
typedef __attribute__((ext_vector_type(8)))  float  v8f;

// ---------------------------------------------------------------------------
// Prep: fp32 -> bf16 conversion of activations
// ---------------------------------------------------------------------------
__global__ void prep_x_kernel(const float* __restrict__ x, bf16* __restrict__ o, int total4) {
    int stride = gridDim.x * blockDim.x;
    for (int i = blockIdx.x * blockDim.x + threadIdx.x; i < total4; i += stride) {
        float4 v = ((const float4*)x)[i];
        v4bf r;
        r[0] = (bf16)v.x; r[1] = (bf16)v.y; r[2] = (bf16)v.z; r[3] = (bf16)v.w;
        ((v4bf*)o)[i] = r;
    }
}

// Prep: ternarize STE-binary weight pair into bf16 {-1,0,+1}
// forward of _ste_binary(w) == (w > 0) ? 1 : 0 ; combined = pos - neg
__global__ void prep_w_kernel(const float* __restrict__ wp, const float* __restrict__ wn,
                              bf16* __restrict__ w, int total4) {
    int stride = gridDim.x * blockDim.x;
    for (int i = blockIdx.x * blockDim.x + threadIdx.x; i < total4; i += stride) {
        float4 p = ((const float4*)wp)[i];
        float4 n = ((const float4*)wn)[i];
        v4bf r;
        r[0] = (bf16)((p.x > 0.f ? 1.f : 0.f) - (n.x > 0.f ? 1.f : 0.f));
        r[1] = (bf16)((p.y > 0.f ? 1.f : 0.f) - (n.y > 0.f ? 1.f : 0.f));
        r[2] = (bf16)((p.z > 0.f ? 1.f : 0.f) - (n.z > 0.f ? 1.f : 0.f));
        r[3] = (bf16)((p.w > 0.f ? 1.f : 0.f) - (n.w > 0.f ? 1.f : 0.f));
        ((v4bf*)w)[i] = r;
    }
}

// ---------------------------------------------------------------------------
// Async global -> LDS copy (CDNA5): each active lane moves 16B directly into
// LDS, tracked by ASYNCcnt. vdst = per-lane LDS byte address, vaddr = 32-bit
// byte offset, saddr = uniform 64-bit base.
// ---------------------------------------------------------------------------
__device__ __forceinline__ void async_b128(unsigned lds_addr, unsigned gofs, const void* base) {
    asm volatile("global_load_async_to_lds_b128 %0, %1, %2"
                 :: "v"(lds_addr), "v"(gofs), "s"((unsigned long long)(size_t)base)
                 : "memory");
}

// ---------------------------------------------------------------------------
// Fragment assembly from two 16B chunks (contiguous k = 16*kh + s, same order for A and B).
// A (16x32 MxK, 16-bit): lane m = lane&15; lanes 0-15 hold K 0..15,
//                        lanes 16-31 hold K 16..31  -> chunks at
//                        row + kh*16 and row + kh*16 + 8 (direct from global).
// B (32x16 KxN, 16-bit): lane n = lane&15; lanes 0-15 hold K 0..15,
//                        lanes 16-31 hold K 16..31 -> chunks at kh*16, kh*16+8.
// ---------------------------------------------------------------------------
__device__ __forceinline__ v16bf load_frag16(const bf16* p0, const bf16* p1) {
    v8bf a = *(const v8bf*)p0;
    v8bf b = *(const v8bf*)p1;
    return __builtin_shufflevector(a, b, 0,1,2,3,4,5,6,7,8,9,10,11,12,13,14,15);
}

__device__ __forceinline__ v8f wmma_bf16(v16bf a, v16bf b, v8f c) {
    // (neg_a, A, neg_b, B, c_mod, C, reuse_a, reuse_b)
    return __builtin_amdgcn_wmma_f32_16x16x32_bf16(false, a, false, b, (short)0, c,
                                                   false, false);
}

// ---------------------------------------------------------------------------
// Fused complex-structured ternary GEMM:
//   y_real    = xr * wr^T + xc * wc^T
//   y_complex = xr * wc^T + xc * wr^T
// Block = 128x128 tile of BOTH outputs; 8 waves in a 4m x 2n grid, each wave
// owns a 32x64 sub-tile = 2x4 wmma tiles x 2 accumulator sets (32 WMMA/stage).
// Weights: double-buffered LDS via async-to-LDS copies (ASYNCcnt).
// Activations: register double-buffered direct global b128 fragment loads.
// ---------------------------------------------------------------------------
__global__ __launch_bounds__(256)
void plinear_complex_gemm(const bf16* __restrict__ xr, const bf16* __restrict__ xc,
                          const bf16* __restrict__ wr, const bf16* __restrict__ wc,
                          float* __restrict__ y_real, float* __restrict__ y_cplx) {
    __shared__ __align__(16) bf16 s_wr[2][BN * LDS_STRIDE];
    __shared__ __align__(16) bf16 s_wc[2][BN * LDS_STRIDE];

    const int tid  = threadIdx.x;
    const int lane = tid & 31;
    const int wave = tid >> 5;     // 0..7
    const int wm   = wave & 3;     // wave row: 32 output rows each
    const int wn   = wave >> 2;    // wave col: 64 output cols each
    const int l15  = lane & 15;
    const int kh   = lane >> 4;

    const int m0 = blockIdx.y * BM;
    const int n0 = blockIdx.x * BN;

    // ---- weight staging slots: 128 rows x 4 x 16B chunks = 512 -> 2/thread ----
    const int w0_row = tid >> 2;
    const int w1_row = (tid + 256) >> 2;
    const int w_cp   = (tid & 3) * 8;
    const unsigned gw0 = (unsigned)(((n0 + w0_row) * KDIM + w_cp) * 2);
    const unsigned gw1 = (unsigned)(((n0 + w1_row) * KDIM + w_cp) * 2);

    unsigned lwr0[2], lwr1[2], lwc0[2], lwc1[2];
#pragma unroll
    for (int b = 0; b < 2; ++b) {
        lwr0[b] = (unsigned)(size_t)&s_wr[b][w0_row * LDS_STRIDE + w_cp];
        lwr1[b] = (unsigned)(size_t)&s_wr[b][w1_row * LDS_STRIDE + w_cp];
        lwc0[b] = (unsigned)(size_t)&s_wc[b][w0_row * LDS_STRIDE + w_cp];
        lwc1[b] = (unsigned)(size_t)&s_wc[b][w1_row * LDS_STRIDE + w_cp];
    }

    // ---- A row base pointers [tensor][i] ----
    const bf16* arow[2][2];
#pragma unroll
    for (int i = 0; i < 2; ++i) {
        size_t r = (size_t)(m0 + wm * 32 + i * 16 + l15);
        arow[0][i] = xr + r * KDIM;
        arow[1][i] = xc + r * KDIM;
    }

    v8f acc_r[2][4], acc_c[2][4];
    const v8f vzero = {0.f, 0.f, 0.f, 0.f, 0.f, 0.f, 0.f, 0.f};
#pragma unroll
    for (int i = 0; i < 2; ++i)
#pragma unroll
        for (int j = 0; j < 4; ++j) { acc_r[i][j] = vzero; acc_c[i][j] = vzero; }

#define ISSUE_W(nb, kbytes) do {                                   \
        async_b128(lwr0[nb], gw0 + (kbytes), wr);                  \
        async_b128(lwr1[nb], gw1 + (kbytes), wr);                  \
        async_b128(lwc0[nb], gw0 + (kbytes), wc);                  \
        async_b128(lwc1[nb], gw1 + (kbytes), wc);                  \
    } while (0)

#define LOAD_A(dst, kk) do {                                       \
        _Pragma("unroll")                                          \
        for (int t_ = 0; t_ < 2; ++t_) {                           \
            _Pragma("unroll")                                      \
            for (int i_ = 0; i_ < 2; ++i_) {                       \
                const bf16* rp_ = arow[t_][i_] + (kk);             \
                dst[t_][i_] = load_frag16(rp_ + kh * 16,           \
                                          rp_ + kh * 16 + 8);      \
            }                                                      \
        }                                                          \
    } while (0)

#define COMPUTE(bufc, A) do {                                                  \
        v16bf bwr_[4], bwc_[4];                                                \
        _Pragma("unroll")                                                      \
        for (int j_ = 0; j_ < 4; ++j_) {                                       \
            int rb_ = (wn * 64 + j_ * 16 + l15) * LDS_STRIDE + kh * 16;        \
            bwr_[j_] = load_frag16(&s_wr[bufc][rb_], &s_wr[bufc][rb_ + 8]);    \
            bwc_[j_] = load_frag16(&s_wc[bufc][rb_], &s_wc[bufc][rb_ + 8]);    \
        }                                                                      \
        _Pragma("unroll")                                                      \
        for (int i_ = 0; i_ < 2; ++i_) {                                       \
            _Pragma("unroll")                                                  \
            for (int j_ = 0; j_ < 4; ++j_) {                                   \
                acc_r[i_][j_] = wmma_bf16(A[0][i_], bwr_[j_], acc_r[i_][j_]);  \
                acc_r[i_][j_] = wmma_bf16(A[1][i_], bwc_[j_], acc_r[i_][j_]);  \
                acc_c[i_][j_] = wmma_bf16(A[0][i_], bwc_[j_], acc_c[i_][j_]);  \
                acc_c[i_][j_] = wmma_bf16(A[1][i_], bwr_[j_], acc_c[i_][j_]);  \
            }                                                                  \
        }                                                                      \
    } while (0)

    // ---- prologue: A frags for stage 0, async weights stage 0 -> buf 0 ----
    v16bf a0[2][2], a1[2][2];
    LOAD_A(a0, 0);
    ISSUE_W(0, 0u);

    for (int s = 0; s < KSTEPS; s += 2) {
        // ---------- even stage s: buf 0, frags a0 ----------
        {
            // s+1 < KSTEPS always (KSTEPS even)
            const unsigned kb = (unsigned)((s + 1) * BK * 2);
            LOAD_A(a1, (s + 1) * BK);          // register prefetch (LOADcnt)
            ISSUE_W(1, kb);                    // async prefetch (ASYNCcnt)
            asm volatile("s_wait_asynccnt 0x4" ::: "memory");  // drain stage s only
            __syncthreads();
            COMPUTE(0, a0);
            __syncthreads();
        }
        // ---------- odd stage s+1: buf 1, frags a1 ----------
        {
            if (s + 2 < KSTEPS) {
                const unsigned kb = (unsigned)((s + 2) * BK * 2);
                LOAD_A(a0, (s + 2) * BK);
                ISSUE_W(0, kb);
                asm volatile("s_wait_asynccnt 0x4" ::: "memory");
            } else {
                asm volatile("s_wait_asynccnt 0x0" ::: "memory");
            }
            __syncthreads();
            COMPUTE(1, a1);
            __syncthreads();
        }
    }

    // ---- epilogue: C/D layout -> global (M = v + 8*kh, N = lane&15) ----
#pragma unroll
    for (int i = 0; i < 2; ++i) {
#pragma unroll
        for (int j = 0; j < 4; ++j) {
            int gcol  = n0 + wn * 64 + j * 16 + l15;
            int growb = m0 + wm * 32 + i * 16 + kh * 8;
#pragma unroll
            for (int v = 0; v < 8; ++v) {
                size_t off = (size_t)(growb + v) * ODIM + gcol;
                y_real[off] = acc_r[i][j][v];
                y_cplx[off] = acc_c[i][j][v];
            }
        }
    }
#undef ISSUE_W
#undef LOAD_A
#undef COMPUTE
}

// ---------------------------------------------------------------------------
// Launch
// ---------------------------------------------------------------------------
extern "C" void kernel_launch(void* const* d_in, const int* in_sizes, int n_in,
                              void* d_out, int out_size, void* d_ws, size_t ws_size,
                              hipStream_t stream) {
    const float* x_real        = (const float*)d_in[0];
    const float* x_complex     = (const float*)d_in[1];
    const float* w_real_pos    = (const float*)d_in[2];
    const float* w_real_neg    = (const float*)d_in[3];
    const float* w_complex_pos = (const float*)d_in[4];
    const float* w_complex_neg = (const float*)d_in[5];

    // workspace layout (bf16): xr[8192x4096] | xc[8192x4096] | wr[4096x4096] | wc[4096x4096]
    bf16* xr_bf = (bf16*)d_ws;
    bf16* xc_bf = xr_bf + (size_t)NROWS * KDIM;
    bf16* wr_bf = xc_bf + (size_t)NROWS * KDIM;
    bf16* wc_bf = wr_bf + (size_t)ODIM * KDIM;

    prep_x_kernel<<<2048, 256, 0, stream>>>(x_real,    xr_bf, (NROWS * KDIM) / 4);
    prep_x_kernel<<<2048, 256, 0, stream>>>(x_complex, xc_bf, (NROWS * KDIM) / 4);
    prep_w_kernel<<<2048, 256, 0, stream>>>(w_real_pos,    w_real_neg,    wr_bf, (ODIM * KDIM) / 4);
    prep_w_kernel<<<2048, 256, 0, stream>>>(w_complex_pos, w_complex_neg, wc_bf, (ODIM * KDIM) / 4);

    float* y_real = (float*)d_out;
    float* y_cplx = y_real + (size_t)NROWS * ODIM;

    dim3 grid(ODIM / BN, NROWS / BM);   // 32 x 64 blocks
    plinear_complex_gemm<<<grid, 256, 0, stream>>>(xr_bf, xc_bf, wr_bf, wc_bf, y_real, y_cplx);
}
